// PointNetFPModule_70153995813277
// MI455X (gfx1250) — matrix operand order repaired and checked by reading, not compiled, not run on hardware
//
#include <hip/hip_runtime.h>
#include <hip/hip_bf16.h>
#include <math.h>

typedef __attribute__((ext_vector_type(16))) _Float16 v16h;
typedef __attribute__((ext_vector_type(8)))  _Float16 v8h;
typedef __attribute__((ext_vector_type(8)))  float    v8f;

#define BB   4
#define NPT  16384
#define MM   2048
#define CF   256
#define CP   64
#define CT   64
#define K0   320
#define K1   256
#define CO   256
#define GRP  8
#define CPG  32
#define GN_EPS 1e-5f

// Tensor Data Mover path (gfx1250): probe-confirmed builtin, arity differs per toolchain.
#if defined(__HIP_DEVICE_COMPILE__) && __has_builtin(__builtin_amdgcn_tensor_load_to_lds) && \
    __has_builtin(__builtin_amdgcn_s_wait_tensorcnt)
#define HAVE_TDM 1
#else
#define HAVE_TDM 0
#endif

#if HAVE_TDM
typedef __attribute__((ext_vector_type(4))) unsigned int u32x4;
typedef __attribute__((ext_vector_type(4))) int          i32x4;
typedef __attribute__((ext_vector_type(8))) int          i32x8;

// Build D# group1 for a 2D tensor (data_size = 2 bytes):
//   dim0/stride0 = innermost (contiguous) extent, dim1 = row count,
//   tile0 x tile1 = tile extent (tile0 along dim0).
__device__ __forceinline__ i32x8 tdm_group1_2d(unsigned dim0, unsigned dim1,
                                               unsigned tile0, unsigned tile1,
                                               unsigned long long stride0) {
  i32x8 g;
  g[0] = (int)(1u << 16);                                   // data_size=1 (2 bytes)
  g[1] = (int)((dim0 & 0xFFFFu) << 16);                     // tensor_dim0[15:0]
  g[2] = (int)((dim0 >> 16) | ((dim1 & 0xFFFFu) << 16));    // dim0[31:16] | dim1[15:0]
  g[3] = (int)((dim1 >> 16) | (tile0 << 16));               // dim1[31:16] | tile_dim0
  g[4] = (int)(tile1);                                      // tile_dim1 | tile_dim2=0
  g[5] = (int)(unsigned)(stride0 & 0xFFFFFFFFull);          // tensor_dim0_stride[31:0]
  g[6] = (int)(unsigned)((stride0 >> 32) & 0xFFFFull);      // stride0[47:32] | stride1 lo=0
  g[7] = 0;                                                 // stride1 hi (unused for 2D)
  return g;
}

__device__ __forceinline__ u32x4 tdm_group0(unsigned lds_addr, const void* gptr) {
  unsigned long long ga = (unsigned long long)(uintptr_t)gptr;
  u32x4 g;
  g[0] = 1u;                                                // count=1, no gather
  g[1] = lds_addr;                                          // LDS byte address
  g[2] = (unsigned)(ga & 0xFFFFFFFFull);                    // global_addr[31:0]
  g[3] = (unsigned)((ga >> 32) & 0x1FFFFFFull) | (2u << 30);// global_addr[56:32] | type=2
  return g;
}

__device__ __forceinline__ void tdm_load_2d(unsigned lds_addr, const void* gptr,
                                            unsigned dim0, unsigned dim1,
                                            unsigned tile0, unsigned tile1,
                                            unsigned long long stride0) {
  u32x4 g0 = tdm_group0(lds_addr, gptr);
  i32x8 g1 = tdm_group1_2d(dim0, dim1, tile0, tile1, stride0);
  i32x4 z4 = {0, 0, 0, 0};
#if __clang_major__ >= 23
  i32x8 z8 = {0, 0, 0, 0, 0, 0, 0, 0};
  __builtin_amdgcn_tensor_load_to_lds(g0, g1, z4, z4, z8, 0);
#else
  __builtin_amdgcn_tensor_load_to_lds(g0, g1, z4, z4, 0);
#endif
}
#endif  // HAVE_TDM

// ---------------- weight fp32 -> fp16 conversion ----------------
__global__ void k_cvt_w(const float* __restrict__ w0, const float* __restrict__ w1,
                        _Float16* __restrict__ w0h, _Float16* __restrict__ w1h) {
  int i = blockIdx.x * 256 + threadIdx.x;
  if (i < CO * K0) w0h[i] = (_Float16)w0[i];
  if (i < CO * K1) w1h[i] = (_Float16)w1[i];
}

// ---------------- nearest-neighbor argmin ----------------
__global__ __launch_bounds__(256) void k_nn(const float* __restrict__ pc,
                                            const float* __restrict__ cc,
                                            int* __restrict__ idx) {
  __shared__ float lc[3 * MM];
  int b = blockIdx.y;
  const float* ccb = cc + (size_t)b * 3 * MM;
  for (int i = threadIdx.x; i < 3 * MM; i += 256) lc[i] = ccb[i];
  __syncthreads();
  int n = blockIdx.x * 256 + threadIdx.x;
  const float* pcb = pc + (size_t)b * 3 * NPT;
  float px = pcb[n], py = pcb[NPT + n], pz = pcb[2 * NPT + n];
  float bd = 3.4e38f;
  int bi = 0;
#pragma unroll 4
  for (int m = 0; m < MM; ++m) {
    float dx = px - lc[m];
    float dy = py - lc[MM + m];
    float dz = pz - lc[2 * MM + m];
    float d = __builtin_fmaf(dx, dx, __builtin_fmaf(dy, dy, dz * dz));
    if (d < bd) { bd = d; bi = m; }
  }
  idx[(size_t)b * NPT + n] = bi;
}

// ---------------- gather + concat -> f16 X; emit temb + coords outputs ----------------
__global__ __launch_bounds__(256) void k_gather(const float* __restrict__ cf,
                                                const float* __restrict__ pf,
                                                const float* __restrict__ tm,
                                                const float* __restrict__ pc,
                                                const int* __restrict__ idx,
                                                _Float16* __restrict__ xb,
                                                float* __restrict__ out_pc,
                                                float* __restrict__ out_tm) {
  int b = blockIdx.y;
  int n = blockIdx.x * 256 + threadIdx.x;
  int j = idx[(size_t)b * NPT + n];
  const float* cfb = cf + (size_t)b * CF * MM;
  _Float16* xbb = xb + (size_t)b * K0 * NPT;
#pragma unroll 4
  for (int c = 0; c < CF; ++c)
    xbb[(size_t)c * NPT + n] = (_Float16)cfb[(size_t)c * MM + j];
  const float* pfb = pf + (size_t)b * CP * NPT;
#pragma unroll 4
  for (int c = 0; c < CP; ++c)
    xbb[(size_t)(CF + c) * NPT + n] = (_Float16)pfb[(size_t)c * NPT + n];
  const float* tmb = tm + (size_t)b * CT * MM;
  float* otb = out_tm + (size_t)b * CT * NPT;
#pragma unroll 4
  for (int c = 0; c < CT; ++c)
    otb[(size_t)c * NPT + n] = tmb[(size_t)c * MM + j];
  const float* pcb = pc + (size_t)b * 3 * NPT;
  float* opb = out_pc + (size_t)b * 3 * NPT;
  for (int d = 0; d < 3; ++d) opb[d * NPT + n] = pcb[d * NPT + n];
}

// ---------------- WMMA GEMM: Y[b,o,n] = sum_k W[o,k] * X[b,k,n] + bias[o] ----------------
// WG tile: 128 out-channels x 128 columns; 8 waves as 4 (ch) x 2 (col).
// Wave tile: 32ch x 64col = 2x4 tiles of v_wmma_f32_16x16x32_f16. K stepped by 32.
// Double-buffered LDS: TDM DMA of tile s+1 overlaps WMMA compute on tile s
// (TENSORcnt completes in order, so waiting for <=2 outstanding means the
// current tile's two loads have landed while the next tile's are in flight).
__global__ __launch_bounds__(256) void k_gemm(const _Float16* __restrict__ W,
                                              const _Float16* __restrict__ X,
                                              const float* __restrict__ bias,
                                              float* __restrict__ Y, int K) {
  __shared__ _Float16 lw[2][128 * 32];   // A tiles: [ch][k]   2 x 8 KB
  __shared__ _Float16 lx[2][32 * 128];   // B tiles: [k][n]    2 x 8 KB
  int b = blockIdx.z;
  int chBase = blockIdx.y * 128;
  int n0 = blockIdx.x * 128;
  const _Float16* Xb = X + (size_t)b * K * NPT;
  float* Yb = Y + (size_t)b * CO * NPT;
  int tid = threadIdx.x;
  int lane = tid & 31, wave = tid >> 5;
  int wr = wave & 3, wc = wave >> 2;
  int r = lane & 15, hi = lane >> 4;
  v8f acc[2][4] = {};

#if HAVE_TDM
  // Pipeline prologue: DMA tile 0 into buffer 0.
  if (wave == 0) {
    tdm_load_2d((unsigned)(uintptr_t)&lw[0][0], &W[(size_t)chBase * K],
                (unsigned)K, (unsigned)CO, 32u, 128u, (unsigned long long)K);
    tdm_load_2d((unsigned)(uintptr_t)&lx[0][0], &Xb[n0],
                (unsigned)NPT, (unsigned)K, 128u, 32u, (unsigned long long)NPT);
  }
#endif

  int buf = 0;
  for (int k0 = 0; k0 < K; k0 += 32, buf ^= 1) {
#if HAVE_TDM
    if (wave == 0) {
      int kn = k0 + 32;
      if (kn < K) {
        // Prefetch tile s+1 into the other buffer (overlaps this step's WMMAs).
        tdm_load_2d((unsigned)(uintptr_t)&lw[buf ^ 1][0],
                    &W[(size_t)chBase * K + kn],
                    (unsigned)K, (unsigned)CO, 32u, 128u, (unsigned long long)K);
        tdm_load_2d((unsigned)(uintptr_t)&lx[buf ^ 1][0],
                    &Xb[(size_t)kn * NPT + n0],
                    (unsigned)NPT, (unsigned)K, 128u, 32u, (unsigned long long)NPT);
        __builtin_amdgcn_s_wait_tensorcnt(2);   // current tile landed; next in flight
      } else {
        __builtin_amdgcn_s_wait_tensorcnt(0);   // drain for the last tile
      }
    }
    __syncthreads();   // current buffer ready for all waves
#else
    {   // cooperative A tile load: 128x32 halfs, 16 halfs/thread
      int t16 = tid * 16;
      int row = t16 >> 5, col = t16 & 31;
      const _Float16* src = &W[(size_t)(chBase + row) * K + k0 + col];
      *(v8h*)&lw[buf][t16]     = *(const v8h*)&src[0];
      *(v8h*)&lw[buf][t16 + 8] = *(const v8h*)&src[8];
    }
    {   // cooperative B tile load: 32x128 halfs, 16 halfs/thread
      int row = tid >> 3, col = (tid & 7) * 16;
      const _Float16* src = &Xb[(size_t)(k0 + row) * NPT + n0 + col];
      _Float16* dst = &lx[buf][row * 128 + col];
      *(v8h*)&dst[0] = *(const v8h*)&src[0];
      *(v8h*)&dst[8] = *(const v8h*)&src[8];
    }
    __syncthreads();
    if (k0 + 32 < K)
      __builtin_prefetch(&Xb[(size_t)(k0 + 32) * NPT + n0 + tid], 0, 1);
#endif

    // A fragments (ISA 16-bit A 16x32 layout): lane r = row,
    // lanes 0-15 hold K 0-7 / 16-23, lanes 16-31 hold K 8-15 / 24-31.
    v16h aa[2];
#pragma unroll
    for (int mi = 0; mi < 2; ++mi) {
      int row = wr * 32 + mi * 16 + r;
      union { v16h v; v8h p[2]; } u;
      u.p[0] = *(const v8h*)&lw[buf][row * 32 + hi * 8];
      u.p[1] = *(const v8h*)&lw[buf][row * 32 + 16 + hi * 8];
      aa[mi] = u.v;
    }
    // B fragments: lane r = column, K order matches A (lanes 0-15 K 0-7 / 16-23, 16-31 K 8-15 / 24-31).
#pragma unroll
    for (int ni = 0; ni < 4; ++ni) {
      v16h bb;
#pragma unroll
      for (int s = 0; s < 16; ++s)
        bb[s] = lx[buf][((s & 7) + hi * 8 + (s >> 3) * 16) * 128 + wc * 64 + ni * 16 + r];
#pragma unroll
      for (int mi = 0; mi < 2; ++mi)
        acc[mi][ni] = __builtin_amdgcn_wmma_f32_16x16x32_f16(
            false, aa[mi], false, bb, (short)0, acc[mi][ni], false, false);
    }
    // End-of-step barrier: nobody may still be reading the buffer that the
    // *next* iteration's DMA (issued by wave 0) will overwrite.
    __syncthreads();
  }

  // C/D layout: col = lane&15, row = v + (lane>=16)*8
#pragma unroll
  for (int mi = 0; mi < 2; ++mi) {
#pragma unroll
    for (int ni = 0; ni < 4; ++ni) {
      int col = n0 + wc * 64 + ni * 16 + r;
      int rowb = chBase + wr * 32 + mi * 16 + hi * 8;
#pragma unroll
      for (int v = 0; v < 8; ++v) {
        int row = rowb + v;
        Yb[(size_t)row * NPT + col] = acc[mi][ni][v] + bias[row];
      }
    }
  }
}

// ---------------- GroupNorm statistics per (batch, group) ----------------
__global__ __launch_bounds__(256) void k_stats(const float* __restrict__ Y,
                                               float* __restrict__ stats) {
  __shared__ float s1[256], s2[256];
  int bg = blockIdx.x;
  int b = bg >> 3, g = bg & 7;
  const float* p = Y + (size_t)b * CO * NPT + (size_t)g * CPG * NPT;
  const int total = CPG * NPT;
  float sum = 0.f, sq = 0.f;
  for (int i = threadIdx.x; i < total; i += 256) {
    float v = p[i];
    sum += v;
    sq = __builtin_fmaf(v, v, sq);
  }
  s1[threadIdx.x] = sum;
  s2[threadIdx.x] = sq;
  __syncthreads();
  for (int s = 128; s > 0; s >>= 1) {
    if (threadIdx.x < s) {
      s1[threadIdx.x] += s1[threadIdx.x + s];
      s2[threadIdx.x] += s2[threadIdx.x + s];
    }
    __syncthreads();
  }
  if (threadIdx.x == 0) {
    float mu = s1[0] / (float)total;
    float var = s2[0] / (float)total - mu * mu;
    stats[bg * 2] = mu;
    stats[bg * 2 + 1] = rsqrtf(var + GN_EPS);
  }
}

// ---------------- normalize + Swish; write f16 (layer1 feed) or fp32 (final) ----------------
__global__ __launch_bounds__(256) void k_norm(const float* __restrict__ Y,
                                              const float* __restrict__ stats,
                                              const float* __restrict__ gamma,
                                              const float* __restrict__ beta,
                                              _Float16* __restrict__ xh,
                                              float* __restrict__ xf,
                                              int write_half) {
  int b = blockIdx.z, c = blockIdx.y;
  int n = blockIdx.x * 256 + threadIdx.x;
  size_t off = (size_t)b * CO * NPT + (size_t)c * NPT + n;
  int g = c >> 5;
  float mu = stats[(b * GRP + g) * 2];
  float rs = stats[(b * GRP + g) * 2 + 1];
  float y = __builtin_fmaf((Y[off] - mu) * rs, gamma[c], beta[c]);
  float s = y / (1.0f + __expf(-y));
  if (write_half) xh[off] = (_Float16)s;
  else            xf[off] = s;
}

extern "C" void kernel_launch(void* const* d_in, const int* in_sizes, int n_in,
                              void* d_out, int out_size, void* d_ws, size_t ws_size,
                              hipStream_t stream) {
  const float* pc  = (const float*)d_in[0];   // [B,3,N]
  const float* cc  = (const float*)d_in[1];   // [B,3,M]
  const float* cf  = (const float*)d_in[2];   // [B,256,M]
  const float* pf  = (const float*)d_in[3];   // [B,64,N]
  const float* tm  = (const float*)d_in[4];   // [B,64,M]
  const float* w0  = (const float*)d_in[5];
  const float* b0  = (const float*)d_in[6];
  const float* g0  = (const float*)d_in[7];
  const float* be0 = (const float*)d_in[8];
  const float* w1  = (const float*)d_in[9];
  const float* b1  = (const float*)d_in[10];
  const float* g1  = (const float*)d_in[11];
  const float* be1 = (const float*)d_in[12];

  // d_out: x [4,256,16384] | points_coords [4,3,16384] | interp_temb [4,64,16384]
  float* outX  = (float*)d_out;
  float* outPC = outX + (size_t)BB * CO * NPT;
  float* outTM = outPC + (size_t)BB * 3 * NPT;

  // workspace layout
  char* ws = (char*)d_ws;
  int*      idx   = (int*)ws;                              // 256 KB
  _Float16* w0h   = (_Float16*)(ws + 262144);              // 160 KB
  _Float16* w1h   = (_Float16*)(ws + 425984);              // 128 KB
  float*    stats = (float*)(ws + 557056);                 // 256 B
  _Float16* xb    = (_Float16*)(ws + (1 << 20));           // 40 MB: [B,320,N] f16

  // 1) weights -> f16
  k_cvt_w<<<(CO * K0 + 255) / 256, 256, 0, stream>>>(w0, w1, w0h, w1h);
  // 2) nearest-neighbor index
  k_nn<<<dim3(NPT / 256, BB), 256, 0, stream>>>(pc, cc, idx);
  // 3) gather + concat + passthrough outputs
  k_gather<<<dim3(NPT / 256, BB), 256, 0, stream>>>(cf, pf, tm, pc, idx, xb, outPC, outTM);
  // 4) layer0 GEMM (WMMA, TDM double-buffered) -> fp32 scratch in outX
  k_gemm<<<dim3(NPT / 128, CO / 128, BB), 256, 0, stream>>>(w0h, xb, b0, outX, K0);
  // 5) GN stats + normalize/Swish -> f16 X1 (reuses xb)
  k_stats<<<BB * GRP, 256, 0, stream>>>(outX, stats);
  k_norm<<<dim3(NPT / 256, CO, BB), 256, 0, stream>>>(outX, stats, g0, be0, xb, outX, 1);
  // 6) layer1 GEMM (WMMA, TDM double-buffered) -> fp32 in outX
  k_gemm<<<dim3(NPT / 128, CO / 128, BB), 256, 0, stream>>>(w1h, xb, b1, outX, K1);
  // 7) GN stats + normalize/Swish in place (final fp32 output)
  k_stats<<<BB * GRP, 256, 0, stream>>>(outX, stats);
  k_norm<<<dim3(NPT / 256, CO, BB), 256, 0, stream>>>(outX, stats, g1, be1, xb, outX, 0);
}
